// MultiHeadAttention_67181878444571
// MI455X (gfx1250) — matrix operand order repaired and checked by reading, not compiled, not run on hardware
//
#include <hip/hip_runtime.h>
#include <hip/hip_bf16.h>

// Problem constants (match reference)
#define BB 2
#define SS 2048
#define DD 1024
#define HH 16
#define DK 64
#define MM (BB*SS)        // 4096 rows
#define NEGV (-1000000000.0f)

typedef __attribute__((ext_vector_type(16))) _Float16 v16h;
typedef __attribute__((ext_vector_type(8)))  _Float16 v8h;
typedef __attribute__((ext_vector_type(8)))  float    v8f;

union F32x8   { v8f  v; float    e[8];  };

// D = A(16x32 f16) * B(32x16 f16) + C, f32 accumulate (codegen-verified builtin)
__device__ __forceinline__ v8f wmma_f16(v16h a, v16h b, v8f c) {
  return __builtin_amdgcn_wmma_f32_16x16x32_f16(false, a, false, b, (short)0, c,
                                                false, false);
}

// A-fragment (16x32, f16, row-major source; `tile` already offset to (m0,k0)).
// Contiguous k order (same as load_b_frag): lanes 0-15 -> K 0..15; lanes 16-31 -> K 16..31.
// Works for both global and LDS (generic) pointers.
__device__ __forceinline__ v16h load_a_frag(const _Float16* tile, int lda,
                                            int mrow, int khalf) {
  const _Float16* p = tile + mrow * lda + 16 * khalf;
  union { v16h v; v8h h[2]; } u;
  u.h[0] = *(const v8h*)(p);
  u.h[1] = *(const v8h*)(p + 8);
  return u.v;
}

// B-fragment (32x16, f16) from an N-major ("transposed") source: Bt[n][k].
// lanes 0-15: K 0..15 of column n=lane; lanes 16-31: K 16..31 of column n=lane-16.
__device__ __forceinline__ v16h load_b_frag(const _Float16* tileT, int ldb,
                                            int nrow, int khalf) {
  return *(const v16h*)(tileT + nrow * ldb + 16 * khalf);
}

// Stage a contiguous 32 KB (16 rows x 1024 f16) strip global -> LDS using the
// CDNA5 async DMA path (GLOBAL_LOAD_ASYNC_TO_LDS_B128, tracked by ASYNCcnt).
// 256 threads x 8 iterations x 16 B. Caller must __syncthreads() after.
__device__ __forceinline__ void async_stage_32k(const _Float16* __restrict__ gsrc,
                                                _Float16* ldst) {
  const char* g = (const char*)gsrc;
  char*       l = (char*)ldst;   // generic pointer to LDS; addr[31:0] = LDS offset
  for (int i = threadIdx.x; i < 2048; i += 256) {
    const uint32_t la = (uint32_t)(size_t)(l + (size_t)i * 16);
    const char*    ga = g + (size_t)i * 16;
    asm volatile("global_load_async_to_lds_b128 %0, %1, off"
                 :: "v"(la), "v"(ga) : "memory");
  }
  asm volatile("s_wait_asynccnt 0x0" ::: "memory");
}

// ---------------------------------------------------------------------------
// Kernel 1: f32 -> f16 conversions; weights transposed so WMMA B loads are
// contiguous per lane.
// ---------------------------------------------------------------------------
__global__ void mha_convert_kernel(const float* __restrict__ x,
                                   const float* __restrict__ wq,
                                   const float* __restrict__ wk,
                                   const float* __restrict__ wv,
                                   const float* __restrict__ wf,
                                   _Float16* __restrict__ xh,
                                   _Float16* __restrict__ wqT,
                                   _Float16* __restrict__ wkT,
                                   _Float16* __restrict__ wvT,
                                   _Float16* __restrict__ wfT) {
  const int tid    = blockIdx.x * blockDim.x + threadIdx.x;
  const int stride = gridDim.x * blockDim.x;
  for (int i = tid; i < MM * DD; i += stride) xh[i] = (_Float16)x[i];
  for (int i = tid; i < DD * DD; i += stride) {
    const int k = i >> 10, n = i & (DD - 1);
    const int t = n * DD + k;          // transpose: wT[n][k] = w[k][n]
    wqT[t] = (_Float16)wq[i];
    wkT[t] = (_Float16)wk[i];
    wvT[t] = (_Float16)wv[i];
    wfT[t] = (_Float16)wf[i];
  }
}

// ---------------------------------------------------------------------------
// Kernel 2: QKV projection GEMM (4096x1024 @ 1024x1024), WMMA f16.
// grid = (MM/16, DD/128, 3); block = 256 (8 waves); wave w owns one 16-col tile.
// The 16-row A strip (32 KB) is staged once into LDS by async DMA; all 8 waves
// feed A fragments from LDS (8x less global A traffic).
// Q stored [b,h,s,dk] pre-scaled by 1/sqrt(DK); K stored [b,h,s,dk];
// V stored transposed [b,h,dk,s] so attention phase-3 B fragments are contiguous.
// ---------------------------------------------------------------------------
__global__ void mha_qkv_kernel(const _Float16* __restrict__ xh,
                               const _Float16* __restrict__ wqT,
                               const _Float16* __restrict__ wkT,
                               const _Float16* __restrict__ wvT,
                               const float* __restrict__ bq,
                               const float* __restrict__ bk,
                               const float* __restrict__ bv,
                               _Float16* __restrict__ Q,
                               _Float16* __restrict__ K,
                               _Float16* __restrict__ Vt) {
  __shared__ _Float16 ldsA[16 * DD];            // 32 KB A strip

  const int wave  = threadIdx.x >> 5;
  const int lane  = threadIdx.x & 31;
  const int mrow  = lane & 15;
  const int khalf = (lane >> 4) & 1;
  const int z     = blockIdx.z;                 // 0=Q 1=K 2=V
  const _Float16* wT  = (z == 0) ? wqT : (z == 1) ? wkT : wvT;
  const float*    bia = (z == 0) ? bq  : (z == 1) ? bk  : bv;
  const int m0 = blockIdx.x * 16;
  const int n0 = blockIdx.y * 128 + wave * 16;

  async_stage_32k(xh + (size_t)m0 * DD, ldsA);
  __syncthreads();

  v8f acc = {};
  for (int k0 = 0; k0 < DD; k0 += 32) {
    const v16h a = load_a_frag(ldsA + k0, DD, mrow, khalf);
    const _Float16* bp = wT + (size_t)n0 * DD + k0;
    __builtin_prefetch(bp + 64, 0, 1);          // global_prefetch_b8
    const v16h b = load_b_frag(bp, DD, mrow, khalf);
    acc = wmma_f16(a, b, acc);
  }

  const int   col = n0 + mrow;
  const float bi  = bia[col];
  const int   h   = col >> 6, dk = col & 63;
  F32x8 c; c.v = acc;
#pragma unroll
  for (int r = 0; r < 8; ++r) {
    const int gr = m0 + r + 8 * khalf;          // global row in [0,4096)
    const int b_ = gr >> 11, s = gr & (SS - 1);
    const float val = c.e[r] + bi;
    if (z == 0)
      Q[((size_t)(b_ * HH + h) * SS + s) * DK + dk] = (_Float16)(val * 0.125f);
    else if (z == 1)
      K[((size_t)(b_ * HH + h) * SS + s) * DK + dk] = (_Float16)val;
    else
      Vt[((size_t)(b_ * HH + h) * DK + dk) * SS + s] = (_Float16)val;
  }
}

// ---------------------------------------------------------------------------
// Kernel 3: fused attention per (b,h, 16-query tile).
//   phase 1: scores = (Q/sqrt(DK)) K^T via WMMA -> 16x2048 f32 strip in LDS
//   phase 2: mask already applied; rowwise softmax (wave shfl reductions),
//            coalesced f32 store of `att` to HBM (the bandwidth limiter)
//   phase 3: ctx = P * V via WMMA (K-dim split over waves, LDS combine)
// Dynamic LDS: 16*2048*4 + 16*64*4 = 135168 B (fits 320 KB WGP LDS).
// ---------------------------------------------------------------------------
__global__ void mha_attn_kernel(const _Float16* __restrict__ Q,
                                const _Float16* __restrict__ K,
                                const _Float16* __restrict__ Vt,
                                const int* __restrict__ mask,
                                float* __restrict__ att,
                                _Float16* __restrict__ ctxh) {
  extern __shared__ float smem[];
  float* scores = smem;                // 16 x 2048
  float* ctxbuf = smem + 16 * SS;      // 16 x 64

  const int wave  = threadIdx.x >> 5;
  const int lane  = threadIdx.x & 31;
  const int mrow  = lane & 15;
  const int khalf = (lane >> 4) & 1;
  const int q0    = blockIdx.x * 16;
  const int bh    = blockIdx.y;        // 0..31
  const int b_    = bh >> 4;

  // Q fragments for this 16-row tile (reused for all 128 key tiles)
  const _Float16* Qb = Q + ((size_t)bh * SS + q0) * DK;
  const v16h a0 = load_a_frag(Qb,      DK, mrow, khalf);   // K-dim 0..31
  const v16h a1 = load_a_frag(Qb + 32, DK, mrow, khalf);   // K-dim 32..63

  // ---- phase 1: scores -> LDS (wave w handles key tiles w, w+8, ...) ----
  for (int kt = wave; kt < SS / 16; kt += 8) {
    const _Float16* Kb = K + ((size_t)bh * SS + kt * 16) * DK;
    const v16h b0 = load_b_frag(Kb,      DK, mrow, khalf);
    const v16h b1 = load_b_frag(Kb + 32, DK, mrow, khalf);
    v8f acc = {};
    acc = wmma_f16(a0, b0, acc);
    acc = wmma_f16(a1, b1, acc);

    const int key = kt * 16 + mrow;                 // this lane's key column
    const int mv  = mask[b_ * SS + key];
    F32x8 c; c.v = acc;
#pragma unroll
    for (int r = 0; r < 8; ++r) {
      const float v = (mv == 0) ? NEGV : c.e[r];
      scores[(r + 8 * khalf) * SS + key] = v;
    }
  }
  __syncthreads();

  // ---- phase 2: softmax rows (wave w -> rows 2w, 2w+1), write att ----
#pragma unroll
  for (int rr = 0; rr < 2; ++rr) {
    const int row = wave * 2 + rr;
    float* srow = scores + row * SS;
    float mx = -3.4e38f;
    for (int c = lane; c < SS; c += 32) mx = fmaxf(mx, srow[c]);
    for (int off = 16; off; off >>= 1) mx = fmaxf(mx, __shfl_xor(mx, off, 32));
    float sum = 0.0f;
    for (int c = lane; c < SS; c += 32) {
      const float e = __expf(srow[c] - mx);
      srow[c] = e;
      sum += e;
    }
    for (int off = 16; off; off >>= 1) sum += __shfl_xor(sum, off, 32);
    const float inv = 1.0f / sum;
    float* arow = att + ((size_t)bh * SS + q0 + row) * SS;
    for (int c = lane; c < SS; c += 32) {
      const float p = srow[c] * inv;
      srow[c] = p;
      arow[c] = p;                                  // coalesced HBM store
    }
  }
  __syncthreads();

  // ---- phase 3: ctx = P(16x2048) x V(2048x64) ----
  const int nt = wave & 3;        // dk tile 0..3
  const int ks = wave >> 2;       // K-dim half 0/1
  const _Float16* Vb = Vt + ((size_t)bh * DK + nt * 16 + mrow) * SS;
  v8f acc = {};
  for (int kt = 0; kt < 32; ++kt) {
    const int kbase = ks * 1024 + kt * 32;
    union { v16h v; _Float16 e[16]; } a;
    const float* sr = scores + mrow * SS + kbase + 16 * khalf;
#pragma unroll
    for (int j = 0; j < 8; ++j) a.e[j]     = (_Float16)sr[j];
#pragma unroll
    for (int j = 0; j < 8; ++j) a.e[j + 8] = (_Float16)sr[j + 8];
    const v16h b = *(const v16h*)(Vb + kbase + 16 * khalf);
    acc = wmma_f16(a.v, b, acc);
  }
  F32x8 c; c.v = acc;
  if (ks == 0) {
#pragma unroll
    for (int r = 0; r < 8; ++r)
      ctxbuf[(r + 8 * khalf) * DK + nt * 16 + mrow] = c.e[r];
  }
  __syncthreads();
  if (ks == 1) {
#pragma unroll
    for (int r = 0; r < 8; ++r)
      ctxbuf[(r + 8 * khalf) * DK + nt * 16 + mrow] += c.e[r];
  }
  __syncthreads();

  // write ctx tile as f16, merged-head layout [b, s, h*64+dk]
  const int h = bh & 15;
  for (int idx = threadIdx.x; idx < 16 * DK; idx += 256) {
    const int row = idx >> 6, dk = idx & 63;
    ctxh[(size_t)(b_ * SS + q0 + row) * DD + h * DK + dk] =
        (_Float16)ctxbuf[idx];
  }
}

// ---------------------------------------------------------------------------
// Kernel 4: out = LayerNorm(ctx @ wf + bf + x). WMMA GEMM (each wave owns a
// 128-col strip of a 16-row tile); the A strip is async-DMA staged into LDS;
// rows staged in 64 KB dynamic LDS, wave-reduced LN.
// ---------------------------------------------------------------------------
__global__ void mha_proj_ln_kernel(const _Float16* __restrict__ ctxh,
                                   const _Float16* __restrict__ wfT,
                                   const float* __restrict__ bf,
                                   const float* __restrict__ x,
                                   const float* __restrict__ ln_g,
                                   const float* __restrict__ ln_b,
                                   float* __restrict__ out) {
  __shared__ _Float16 ldsA[16 * DD];   // 32 KB A strip (ctx rows)
  extern __shared__ float ybuf[];      // 16 x 1024
  const int wave  = threadIdx.x >> 5;
  const int lane  = threadIdx.x & 31;
  const int mrow  = lane & 15;
  const int khalf = (lane >> 4) & 1;
  const int m0    = blockIdx.x * 16;

  async_stage_32k(ctxh + (size_t)m0 * DD, ldsA);
  __syncthreads();

  v8f acc[8];
#pragma unroll
  for (int i = 0; i < 8; ++i) acc[i] = (v8f){};

  for (int k0 = 0; k0 < DD; k0 += 32) {
    const v16h a = load_a_frag(ldsA + k0, DD, mrow, khalf);
#pragma unroll
    for (int i = 0; i < 8; ++i) {
      const int n0 = wave * 128 + i * 16;
      const v16h b = load_b_frag(wfT + (size_t)n0 * DD + k0, DD, mrow, khalf);
      acc[i] = wmma_f16(a, b, acc[i]);
    }
  }

#pragma unroll
  for (int i = 0; i < 8; ++i) {
    const int n   = wave * 128 + i * 16 + mrow;
    const float bfv = bf[n];
    F32x8 c; c.v = acc[i];
#pragma unroll
    for (int r = 0; r < 8; ++r) {
      const int m  = r + 8 * khalf;
      const int gr = m0 + m;
      ybuf[m * DD + n] = c.e[r] + bfv + x[(size_t)gr * DD + n];
    }
  }
  __syncthreads();

#pragma unroll
  for (int rr = 0; rr < 2; ++rr) {
    const int row = wave * 2 + rr;
    const float* yr = ybuf + row * DD;
    float s1 = 0.0f, s2 = 0.0f;
    for (int c = lane; c < DD; c += 32) {
      const float y = yr[c];
      s1 += y; s2 += y * y;
    }
    for (int off = 16; off; off >>= 1) {
      s1 += __shfl_xor(s1, off, 32);
      s2 += __shfl_xor(s2, off, 32);
    }
    const float mu   = s1 * (1.0f / DD);
    const float var  = s2 * (1.0f / DD) - mu * mu;
    const float rstd = rsqrtf(var + 1e-5f);
    const int gr = m0 + row;
    for (int c = lane; c < DD; c += 32)
      out[(size_t)gr * DD + c] = (yr[c] - mu) * rstd * ln_g[c] + ln_b[c];
  }
}

// ---------------------------------------------------------------------------
extern "C" void kernel_launch(void* const* d_in, const int* in_sizes, int n_in,
                              void* d_out, int out_size, void* d_ws, size_t ws_size,
                              hipStream_t stream) {
  (void)in_sizes; (void)n_in; (void)out_size;

  const float* x    = (const float*)d_in[0];
  const int*   mask = (const int*)d_in[1];
  const float* wq   = (const float*)d_in[2];
  const float* bq   = (const float*)d_in[3];
  const float* wk   = (const float*)d_in[4];
  const float* bk   = (const float*)d_in[5];
  const float* wv   = (const float*)d_in[6];
  const float* bv   = (const float*)d_in[7];
  const float* wf   = (const float*)d_in[8];
  const float* bf   = (const float*)d_in[9];
  const float* ln_g = (const float*)d_in[10];
  const float* ln_b = (const float*)d_in[11];

  // Workspace layout (f16), total 48 MB
  char* ws = (char*)d_ws;
  size_t off = 0;
  _Float16* xh   = (_Float16*)(ws + off); off += (size_t)MM * DD * 2;
  _Float16* wqT  = (_Float16*)(ws + off); off += (size_t)DD * DD * 2;
  _Float16* wkT  = (_Float16*)(ws + off); off += (size_t)DD * DD * 2;
  _Float16* wvT  = (_Float16*)(ws + off); off += (size_t)DD * DD * 2;
  _Float16* wfT  = (_Float16*)(ws + off); off += (size_t)DD * DD * 2;
  _Float16* Qs   = (_Float16*)(ws + off); off += (size_t)MM * DD * 2;
  _Float16* Kh   = (_Float16*)(ws + off); off += (size_t)MM * DD * 2;
  _Float16* Vt   = (_Float16*)(ws + off); off += (size_t)MM * DD * 2;
  _Float16* ctxh = (_Float16*)(ws + off); off += (size_t)MM * DD * 2;
  if (off > ws_size) return;   // workspace too small: deterministic no-op

  float* out = (float*)d_out;                      // (B,S,D)
  float* att = out + (size_t)MM * DD;              // (B,H,S,S)

  // 1) convert / transpose to f16
  mha_convert_kernel<<<dim3(2048), dim3(256), 0, stream>>>(
      x, wq, wk, wv, wf, xh, wqT, wkT, wvT, wfT);

  // 2) QKV projection (z selects Q/K/V)
  mha_qkv_kernel<<<dim3(MM / 16, DD / 128, 3), dim3(256), 0, stream>>>(
      xh, wqT, wkT, wvT, bq, bk, bv, Qs, Kh, Vt);

  // 3) fused attention (135168 B dynamic LDS per workgroup)
  const int attn_lds = (16 * SS + 16 * DK) * (int)sizeof(float);
  (void)hipFuncSetAttribute((const void*)mha_attn_kernel,
                            hipFuncAttributeMaxDynamicSharedMemorySize, attn_lds);
  mha_attn_kernel<<<dim3(SS / 16, BB * HH), dim3(256), attn_lds, stream>>>(
      Qs, Kh, Vt, mask, att, ctxh);

  // 4) output projection + residual + LayerNorm
  const int proj_lds = 16 * DD * (int)sizeof(float);
  (void)hipFuncSetAttribute((const void*)mha_proj_ln_kernel,
                            hipFuncAttributeMaxDynamicSharedMemorySize, proj_lds);
  mha_proj_ln_kernel<<<dim3(MM / 16), dim3(256), proj_lds, stream>>>(
      ctxh, wfT, bf, x, ln_g, ln_b, out);
}
